// LayerNormLSTMCell_67095979098853
// MI455X (gfx1250) — matrix operand order repaired and checked by reading, not compiled, not run on hardware
//
#include <hip/hip_runtime.h>
#include <math.h>

// ---------------------------------------------------------------------------
// LayerNorm LSTM cell for MI455X (gfx1250, wave32, WMMA)
//   pre = [x|h] (4096x4096) @ [W|U]^T (8192x4096)  -> bf16 WMMA, f32 accum
//   async global->LDS double-buffered staging (ASYNCcnt), then per-row
//   LayerNorm + gates + cell update.
// ---------------------------------------------------------------------------

typedef __attribute__((ext_vector_type(16))) __bf16 v16bf;
typedef __attribute__((ext_vector_type(8)))  __bf16 v8bf;
typedef __attribute__((ext_vector_type(8)))  float  v8f;
typedef int v4i_vec __attribute__((vector_size(16)));   // matches builtin param

#define B_DIM 4096
#define H_DIM 2048
#define K_DIM 4096   // I + H (concatenated contraction dim)
#define N_DIM 8192   // 4 gates * H
#define KSLAB 64
#define LDS_STRIDE 64  // no pad; 2 buffers x (A+B) x 128x64 bf16 = 64 KB LDS

#if __has_builtin(__builtin_amdgcn_global_load_async_to_lds_b128)
#define HAS_ASYNC_LDS 1
#else
#define HAS_ASYNC_LDS 0
#endif

#if __has_builtin(__builtin_amdgcn_s_wait_asynccnt)
#define WAIT_ASYNC(n) __builtin_amdgcn_s_wait_asynccnt(n)
#else
#define WAIT_ASYNC(n) asm volatile("s_wait_asynccnt %0" ::"i"(n) : "memory")
#endif

#define AS1V(p) ((__attribute__((address_space(1))) v4i_vec*)(p))
#define AS3V(p) ((__attribute__((address_space(3))) v4i_vec*)(p))

union FragAB { v16bf v; v8bf h[2]; };

// ---- pack [x | h] row-major -> bf16 ---------------------------------------
__global__ __launch_bounds__(256)
void pack_xh(const float* __restrict__ x, const float* __restrict__ h,
             __bf16* __restrict__ Xc) {
  size_t idx = (size_t)blockIdx.x * 256 + threadIdx.x;   // over B*K
  int    k   = (int)(idx & (K_DIM - 1));
  size_t b   = idx >> 12;
  float  v   = (k < H_DIM) ? x[b * H_DIM + k] : h[b * H_DIM + (k - H_DIM)];
  Xc[idx] = (__bf16)v;
}

// ---- pack gate-stacked [W | U] -> bf16 ------------------------------------
__global__ __launch_bounds__(256)
void pack_w(const float* __restrict__ wi, const float* __restrict__ ui,
            const float* __restrict__ wf, const float* __restrict__ uf,
            const float* __restrict__ wc, const float* __restrict__ uc,
            const float* __restrict__ wo, const float* __restrict__ uo,
            __bf16* __restrict__ Wcat) {
  size_t idx = (size_t)blockIdx.x * 256 + threadIdx.x;   // over N*K
  int k = (int)(idx & (K_DIM - 1));
  int n = (int)(idx >> 12);
  int g = n >> 11;
  int j = n & (H_DIM - 1);
  const float* W; const float* U;
  switch (g) {
    case 0:  W = wi; U = ui; break;
    case 1:  W = wf; U = uf; break;
    case 2:  W = wc; U = uc; break;
    default: W = wo; U = uo; break;
  }
  float v = (k < H_DIM) ? W[(size_t)j * H_DIM + k]
                        : U[(size_t)j * H_DIM + (k - H_DIM)];
  Wcat[idx] = (__bf16)v;
}

__global__ __launch_bounds__(256)
void pack_bias(const float* __restrict__ bi, const float* __restrict__ bf_,
               const float* __restrict__ bc, const float* __restrict__ bo,
               float* __restrict__ biasc) {
  int n = blockIdx.x * 256 + threadIdx.x;
  int g = n >> 11;
  int j = n & (H_DIM - 1);
  biasc[n] = (g == 0) ? bi[j] : (g == 1) ? bf_[j] : (g == 2) ? bc[j] : bo[j];
}

// ---- WMMA GEMM: pre[M=4096][N=8192] = Xc @ Wc^T + bias --------------------
// 128x128 workgroup tile, 8 waves (2x4), each wave 64x32 = 4x2 16x16 tiles.
// K staged through LDS in 64-wide slabs, double-buffered via async copies.
__global__ __launch_bounds__(256)
void gemm_wmma(const __bf16* __restrict__ Xc, const __bf16* __restrict__ Wc,
               const float* __restrict__ bias, float* __restrict__ pre) {
  __shared__ __align__(16) __bf16 As[2][128 * LDS_STRIDE];
  __shared__ __align__(16) __bf16 Bs[2][128 * LDS_STRIDE];

  const int tid   = threadIdx.x;
  const int lane  = tid & 31;
  const int wave  = tid >> 5;
  const int wm    = wave >> 2;    // 0..1  -> which 64-row half
  const int wn    = wave & 3;     // 0..3  -> which 32-col slice
  const int lmod  = lane & 15;
  const int lhalf = lane >> 4;

  const int m0 = blockIdx.y * 128;
  const int n0 = blockIdx.x * 128;

  const v8f vzero = {0.f, 0.f, 0.f, 0.f, 0.f, 0.f, 0.f, 0.f};
  v8f acc[4][2];
  for (int tm = 0; tm < 4; ++tm)
    for (int tn = 0; tn < 2; ++tn)
      acc[tm][tn] = vzero;

  // Stage one 128x64 bf16 slab of A and B into LDS buffer `buf`.
  // 1024 chunks of 8 bf16 (16B) per slab; 4 chunks per thread.
  auto stage = [&](int buf, int k0) {
    for (int i = 0; i < 4; ++i) {
      int c   = tid + i * 256;
      int row = c >> 3;
      int kc  = (c & 7) * 8;
      const __bf16* ga = Xc + (size_t)(m0 + row) * K_DIM + k0 + kc;
      const __bf16* gb = Wc + (size_t)(n0 + row) * K_DIM + k0 + kc;
      __bf16* la = &As[buf][row * LDS_STRIDE + kc];
      __bf16* lb = &Bs[buf][row * LDS_STRIDE + kc];
#if HAS_ASYNC_LDS
      __builtin_amdgcn_global_load_async_to_lds_b128(AS1V(ga), AS3V(la), 0, 0);
      __builtin_amdgcn_global_load_async_to_lds_b128(AS1V(gb), AS3V(lb), 0, 0);
#else
      *(uint4*)la = *(const uint4*)ga;
      *(uint4*)lb = *(const uint4*)gb;
#endif
    }
  };

  constexpr int NK = K_DIM / KSLAB;   // 64 slabs
  stage(0, 0);

  for (int kt = 0; kt < NK; ++kt) {
    const int buf = kt & 1;
    if (kt + 1 < NK) {
      stage(buf ^ 1, (kt + 1) * KSLAB);
#if HAS_ASYNC_LDS
      // 8 loads just issued for the next slab remain outstanding; in-order
      // completion implies this wave's current-slab loads have landed.
      WAIT_ASYNC(8);
#endif
    } else {
#if HAS_ASYNC_LDS
      WAIT_ASYNC(0);
#endif
    }
    __syncthreads();   // everyone's current-slab data resident

    const __bf16* Ab = As[buf];
    const __bf16* Bb = Bs[buf];

    for (int kk = 0; kk < KSLAB; kk += 32) {
      FragAB a[4], bq[2];
      // A-matrix 16x32 bf16 layout: lane l<16 -> row l, K 0..7 / 16..23;
      // lane l>=16 -> row l-16, K 8..15 / 24..31.
      for (int tm = 0; tm < 4; ++tm) {
        const __bf16* p =
            Ab + (wm * 64 + tm * 16 + lmod) * LDS_STRIDE + kk + 8 * lhalf;
        a[tm].h[0] = *(const v8bf*)p;
        a[tm].h[1] = *(const v8bf*)(p + 16);
      }
      // B-matrix 32x16 bf16 layout: lane l -> column l%16,
      // K 0..7 / 16..23 (l<16) or 8..15 / 24..31 (l>=16), matching A.
      for (int tn = 0; tn < 2; ++tn) {
        const __bf16* p =
            Bb + (wn * 32 + tn * 16 + lmod) * LDS_STRIDE + kk + 8 * lhalf;
        bq[tn].h[0] = *(const v8bf*)p;
        bq[tn].h[1] = *(const v8bf*)(p + 16);
      }
      for (int tm = 0; tm < 4; ++tm)
        for (int tn = 0; tn < 2; ++tn)
          acc[tm][tn] = __builtin_amdgcn_wmma_f32_16x16x32_bf16(
              false, a[tm].v, false, bq[tn].v, (short)0, acc[tm][tn],
              false, false);
    }
    __syncthreads();   // all waves done reading `buf` before it is restaged
  }

  // Epilogue: C/D layout -> element (m = r + 8*lhalf, n = lane%16); add bias.
  for (int tm = 0; tm < 4; ++tm)
    for (int tn = 0; tn < 2; ++tn) {
      int   n  = n0 + wn * 32 + tn * 16 + lmod;
      float bv = bias[n];
      for (int r = 0; r < 8; ++r) {
        int m = m0 + wm * 64 + tm * 16 + 8 * lhalf + r;
        pre[(size_t)m * N_DIM + n] = acc[tm][tn][r] + bv;
      }
    }
}

// ---- per-row LayerNorm + gates + cell update ------------------------------
__global__ __launch_bounds__(256)
void ln_lstm(const float* __restrict__ pre, const float* __restrict__ cin,
             const float* __restrict__ g0, const float* __restrict__ b0,
             const float* __restrict__ g1, const float* __restrict__ b1,
             const float* __restrict__ g2, const float* __restrict__ b2,
             const float* __restrict__ g3, const float* __restrict__ b3,
             float* __restrict__ out) {
  __shared__ float s1[256];
  __shared__ float s2[256];
  const int tid = threadIdx.x;
  const int b   = blockIdx.x;
  const float* rowp = pre + (size_t)b * N_DIM;

  const float* gam[4] = {g0, g1, g2, g3};
  const float* bet[4] = {b0, b1, b2, b3};

  float vals[4][8];
  for (int g = 0; g < 4; ++g) {
    float ls = 0.f, lq = 0.f;
    for (int t = 0; t < 8; ++t) {
      float v = rowp[g * H_DIM + t * 256 + tid];
      vals[g][t] = v;
      ls += v;
      lq += v * v;
    }
    s1[tid] = ls;
    s2[tid] = lq;
    __syncthreads();
    for (int off = 128; off > 0; off >>= 1) {
      if (tid < off) {
        s1[tid] += s1[tid + off];
        s2[tid] += s2[tid + off];
      }
      __syncthreads();
    }
    float mean = s1[0] * (1.0f / H_DIM);
    float var  = s2[0] * (1.0f / H_DIM) - mean * mean;
    float inv  = rsqrtf(var + 1e-5f);
    __syncthreads();  // protect s1/s2 before next gate's writes

    for (int t = 0; t < 8; ++t) {
      int   j  = t * 256 + tid;
      float nv = (vals[g][t] - mean) * inv * gam[g][j] + bet[g][j];
      vals[g][t] = (g == 2) ? tanhf(nv) : 1.0f / (1.0f + __expf(-nv));
    }
  }

  for (int t = 0; t < 8; ++t) {
    int   j  = t * 256 + tid;
    float cv = cin[(size_t)b * H_DIM + j];
    float cn = vals[1][t] * cv + vals[0][t] * vals[2][t];
    float hn = vals[3][t] * tanhf(cn);
    out[(size_t)b * H_DIM + j] = hn;                           // h_new
    out[(size_t)B_DIM * H_DIM + (size_t)b * H_DIM + j] = cn;   // c_new
  }
}

// ---------------------------------------------------------------------------
extern "C" void kernel_launch(void* const* d_in, const int* in_sizes, int n_in,
                              void* d_out, int out_size, void* d_ws,
                              size_t ws_size, hipStream_t stream) {
  (void)in_sizes; (void)n_in; (void)out_size; (void)ws_size;

  const float* x = (const float*)d_in[0];
  const float* h = (const float*)d_in[1];
  const float* c = (const float*)d_in[2];
  // per-gate (i,f,c,o): Wg_w, Wg_b, Ug_w, lng_g, lng_b at stride 5 from 3
  const float* Ww[4], *Wb[4], *Uw[4], *lg[4], *lb[4];
  for (int g = 0; g < 4; ++g) {
    Ww[g] = (const float*)d_in[3 + 5 * g + 0];
    Wb[g] = (const float*)d_in[3 + 5 * g + 1];
    Uw[g] = (const float*)d_in[3 + 5 * g + 2];
    lg[g] = (const float*)d_in[3 + 5 * g + 3];
    lb[g] = (const float*)d_in[3 + 5 * g + 4];
  }

  char*   ws    = (char*)d_ws;
  __bf16* Xc    = (__bf16*)ws;                              // 32 MB
  __bf16* Wc    = (__bf16*)(ws + ((size_t)32 << 20));       // 64 MB
  float*  biasc = (float*)(ws + ((size_t)96 << 20));        // 32 KB
  float*  pre   = (float*)(ws + ((size_t)97 << 20));        // 128 MB
  float*  out   = (float*)d_out;

  pack_xh<<<(B_DIM * (size_t)K_DIM) / 256, 256, 0, stream>>>(x, h, Xc);
  pack_w<<<((size_t)N_DIM * K_DIM) / 256, 256, 0, stream>>>(
      Ww[0], Uw[0], Ww[1], Uw[1], Ww[2], Uw[2], Ww[3], Uw[3], Wc);
  pack_bias<<<N_DIM / 256, 256, 0, stream>>>(Wb[0], Wb[1], Wb[2], Wb[3], biasc);

  dim3 grid(N_DIM / 128, B_DIM / 128);
  gemm_wmma<<<grid, 256, 0, stream>>>(Xc, Wc, biasc, pre);

  ln_lstm<<<B_DIM, 256, 0, stream>>>(pre, c, lg[0], lb[0], lg[1], lb[1],
                                     lg[2], lb[2], lg[3], lb[3], out);
}
